// BlockSparseSelfAttention_13391708029369
// MI455X (gfx1250) — matrix operand order repaired and checked by reading, not compiled, not run on hardware
//
#include <hip/hip_runtime.h>
#include <hip/hip_fp16.h>

typedef _Float16 half_t;
typedef __attribute__((ext_vector_type(16))) _Float16 v16h;
typedef __attribute__((ext_vector_type(8)))  _Float16 v8h;
typedef __attribute__((ext_vector_type(8)))  float    v8f;

#define DCH    512          // model dim
#define D3     1536         // 3*D
#define HEADS  8
#define HD     64           // head dim
#define BLK    16           // attention block length
#define MB     4            // 16-token blocks per workgroup
#define MROWS  (MB*BLK)     // 64 rows per workgroup
#define WAVES  8
#define NTHREADS (WAVES*32)

// ---- WMMA operand loaders (contiguous k = 16h + e order, wave32) -------------------
// A 16x32 f16: lane<16 -> M=lane, e=K(k0+e)
//              lane>=16 -> M=lane-16, e=K(k0+16+e)
static __device__ inline v16h loadA16(const half_t* base, int stride, int m0,
                                      int k0, int lane) {
  const int row = m0 + (lane & 15);
  const int ko  = k0 + ((lane >> 4) << 4);
  const half_t* p = base + row * stride + ko;
  v8h lo = *(const v8h*)(p);
  v8h hi = *(const v8h*)(p + 8);
  v16h a;
#pragma unroll
  for (int i = 0; i < 8; ++i) { a[i] = lo[i]; a[8 + i] = hi[i]; }
  return a;
}

// B 32x16 f16 (stored as N-major [N][K]): lane<16 -> N=lane, e=K(k0+e)
//                                         lane>=16 -> N=lane-16, e=K(k0+16+e)
static __device__ inline v16h loadB16(const half_t* base, int stride, int n0,
                                      int k0, int lane) {
  const int row = n0 + (lane & 15);
  const int ko  = k0 + ((lane >> 4) << 4);
  return *(const v16h*)(base + row * stride + ko);
}

static __device__ inline v8f wmma_f16(v16h a, v16h b, v8f c) {
  return __builtin_amdgcn_wmma_f32_16x16x32_f16(false, a, false, b, (short)0,
                                                c, false, false);
}

// ---- prep: fp32 -> f16 weight conversion ---------------------------------
__global__ void cvt_weights(const float* __restrict__ w_in,
                            const float* __restrict__ w_out,
                            half_t* __restrict__ w_in_h,
                            half_t* __restrict__ w_out_h) {
  const int i = blockIdx.x * blockDim.x + threadIdx.x;
  if (i < D3 * DCH)  w_in_h[i]  = (half_t)w_in[i];
  if (i < DCH * DCH) w_out_h[i] = (half_t)w_out[i];
}

// ---- fused block-sparse attention ----------------------------------------
__global__ __launch_bounds__(NTHREADS)
void fused_block_attn(const float* __restrict__ x,
                      const half_t* __restrict__ w_in_h,
                      const float* __restrict__ b_in,
                      const half_t* __restrict__ w_out_h,
                      const float* __restrict__ b_out,
                      float* __restrict__ out) {
  __shared__ half_t lds_x[MROWS * DCH];           // 64 KB: x f16, reused for ctx
  __shared__ half_t lds_qkv[MROWS * D3];          // 192 KB
  __shared__ half_t lds_attn[WAVES][BLK * BLK];   // 4 KB per-wave scratch

  const int tid  = threadIdx.x;
  const int lane = tid & 31;
  const int wave = tid >> 5;
  const int rowbase = blockIdx.x * MROWS;         // global token row

  // ---- stage x block [64 x 512] fp32 -> f16 LDS
  {
    const float4* xg = (const float4*)(x + (size_t)rowbase * DCH);
#pragma unroll 4
    for (int i = tid; i < MROWS * DCH / 4; i += NTHREADS) {
      float4 v = xg[i];
      half_t* p = &lds_x[i * 4];
      p[0] = (half_t)v.x; p[1] = (half_t)v.y;
      p[2] = (half_t)v.z; p[3] = (half_t)v.w;
    }
  }
  __syncthreads();

  // ---- QKV projection: [64x512] @ w_in^T[512x1536] -> lds_qkv (f16)
#pragma unroll 1
  for (int t = 0; t < 12; ++t) {
    const int n0 = (wave * 12 + t) * 16;          // output-feature tile
    v8f acc[MB] = {};
#pragma unroll 1
    for (int kt = 0; kt < DCH / 32; ++kt) {
      v16h bf = loadB16(w_in_h, DCH, n0, kt * 32, lane);
#pragma unroll
      for (int m = 0; m < MB; ++m) {
        v16h af = loadA16(lds_x, DCH, m * 16, kt * 32, lane);
        acc[m] = wmma_f16(af, bf, acc[m]);
      }
    }
    const float bias = b_in[n0 + (lane & 15)];
    const float scl  = (n0 < DCH) ? 0.125f : 1.0f; // fold q * 1/sqrt(64)
#pragma unroll
    for (int m = 0; m < MB; ++m) {
#pragma unroll
      for (int r = 0; r < 8; ++r) {
        const int row = m * 16 + r + ((lane >> 4) << 3);
        lds_qkv[row * D3 + n0 + (lane & 15)] = (half_t)((acc[m][r] + bias) * scl);
      }
    }
  }
  __syncthreads();

  // ---- attention: 32 (block, head) pairs, 4 per wave, fully wave-local
#pragma unroll 1
  for (int i = 0; i < MB * HEADS / WAVES; ++i) {
    const int pair = wave * 4 + i;
    const int blk  = pair >> 3;                   // 0..3
    const int head = pair & 7;
    const half_t* qb = lds_qkv + blk * 16 * D3 + head * HD;
    const half_t* kb = qb + DCH;
    const half_t* vb = qb + 2 * DCH;

    // scores = (q*scale) @ k^T, K = 64 -> two WMMAs
    v8f sc = {};
    sc = wmma_f16(loadA16(qb, D3, 0, 0,  lane), loadB16(kb, D3, 0, 0,  lane), sc);
    sc = wmma_f16(loadA16(qb, D3, 0, 32, lane), loadB16(kb, D3, 0, 32, lane), sc);

    // softmax over N (N = lane%16 -> xor-reduce masks 1,2,4,8)
#pragma unroll
    for (int r = 0; r < 8; ++r) {
      float mv = sc[r];
      mv = fmaxf(mv, __shfl_xor(mv, 8, 32));
      mv = fmaxf(mv, __shfl_xor(mv, 4, 32));
      mv = fmaxf(mv, __shfl_xor(mv, 2, 32));
      mv = fmaxf(mv, __shfl_xor(mv, 1, 32));
      float e = __expf(sc[r] - mv);
      float s = e;
      s += __shfl_xor(s, 8, 32);
      s += __shfl_xor(s, 4, 32);
      s += __shfl_xor(s, 2, 32);
      s += __shfl_xor(s, 1, 32);
      sc[r] = e / s;
    }

    // stash attn (f16) in per-wave scratch, re-read cross-lane in A layout
    half_t* at = &lds_attn[wave][0];
#pragma unroll
    for (int r = 0; r < 8; ++r)
      at[(r + ((lane >> 4) << 3)) * 16 + (lane & 15)] = (half_t)sc[r];
    asm volatile("s_wait_dscnt 0x0" ::: "memory");

    // A = attn padded 16x16 -> 16x32 (upper K half zero)
    v16h af = {};
    {
      const half_t* p = at + (lane & 15) * 16;
#pragma unroll
      if (lane < 16) for (int e = 0; e < 16; ++e) af[e] = p[e];
    }

    // ctx = attn @ v : N = 64 -> 4 tiles; B K rows 16..31 are zero
#pragma unroll 1
    for (int nt = 0; nt < 4; ++nt) {
      v16h bf = {};
      if (lane < 16) {
        const half_t* p = vb + nt * 16 + lane;    // column N = nt*16+lane
#pragma unroll
        for (int e = 0; e < 16; ++e) bf[e] = p[e * D3];  // K = token e
      }
      v8f c = {};
      c = wmma_f16(af, bf, c);
#pragma unroll
      for (int r = 0; r < 8; ++r) {
        const int row = blk * 16 + r + ((lane >> 4) << 3);
        lds_x[row * DCH + head * HD + nt * 16 + (lane & 15)] = (half_t)c[r];
      }
    }
  }
  __syncthreads();

  // ---- output projection: ctx[64x512] @ w_out^T[512x512] + b_out -> out
#pragma unroll 1
  for (int t = 0; t < 4; ++t) {
    const int n0 = (wave * 4 + t) * 16;
    v8f acc[MB] = {};
#pragma unroll 1
    for (int kt = 0; kt < DCH / 32; ++kt) {
      v16h bf = loadB16(w_out_h, DCH, n0, kt * 32, lane);
#pragma unroll
      for (int m = 0; m < MB; ++m) {
        v16h af = loadA16(lds_x, DCH, m * 16, kt * 32, lane);
        acc[m] = wmma_f16(af, bf, acc[m]);
      }
    }
    const float bias = b_out[n0 + (lane & 15)];
#pragma unroll
    for (int m = 0; m < MB; ++m) {
#pragma unroll
      for (int r = 0; r < 8; ++r) {
        const size_t row = (size_t)rowbase + m * 16 + r + ((lane >> 4) << 3);
        out[row * DCH + n0 + (lane & 15)] = acc[m][r] + bias;
      }
    }
  }
}

extern "C" void kernel_launch(void* const* d_in, const int* in_sizes, int n_in,
                              void* d_out, int out_size, void* d_ws, size_t ws_size,
                              hipStream_t stream) {
  const float* x     = (const float*)d_in[0];
  const float* w_in  = (const float*)d_in[1];
  const float* b_in  = (const float*)d_in[2];
  const float* w_out = (const float*)d_in[3];
  const float* b_out = (const float*)d_in[4];

  half_t* w_in_h  = (half_t*)d_ws;                   // 1536*512 f16 = 1.5 MB
  half_t* w_out_h = w_in_h + (size_t)D3 * DCH;       // 512*512  f16 = 0.5 MB

  cvt_weights<<<(D3 * DCH + 255) / 256, 256, 0, stream>>>(w_in, w_out,
                                                          w_in_h, w_out_h);

  const int nwg = (8 * 8192) / MROWS;                // 1024 workgroups
  fused_block_attn<<<nwg, NTHREADS, 0, stream>>>(x, w_in_h, b_in, w_out_h,
                                                 b_out, (float*)d_out);
}
